// NeuralNetwork_85255100825763
// MI455X (gfx1250) — matrix operand order repaired and checked by reading, not compiled, not run on hardware
//
#include <hip/hip_runtime.h>
#include <hip/hip_bf16.h>

// ---------------------------------------------------------------------------
// Shapes (fixed by the reference)
//   B=64, H=W=256, CH=16, NCLS=8, HID=128, ARR=256 (=T of the LSTM scan)
// ---------------------------------------------------------------------------

typedef __attribute__((ext_vector_type(8)))  float   v8f;
typedef __attribute__((ext_vector_type(8)))  __bf16  v8bf;
typedef __attribute__((ext_vector_type(16))) __bf16  v16bf;

__device__ __forceinline__ float sigm(float x) { return 1.f / (1.f + __expf(-x)); }
__device__ __forceinline__ float tanh_fast(float x) { return 2.f / (1.f + __expf(-2.f * x)) - 1.f; }

// ---------------------------------------------------------------------------
// Kernel 1: fully fused conv1+bn1+relu+pool1 + conv2+bn2+relu+pool2 -> c(64,256)
// One thread per c element; every x value is read exactly once globally.
// ---------------------------------------------------------------------------
__global__ __launch_bounds__(256) void fused_conv_kernel(
    const float* __restrict__ x,
    const float* __restrict__ w1, const float* __restrict__ cb1,
    const float* __restrict__ g1, const float* __restrict__ be1,
    const float* __restrict__ m1, const float* __restrict__ v1,
    const float* __restrict__ w2, const float* __restrict__ cb2,
    const float* __restrict__ g2, const float* __restrict__ be2,
    const float* __restrict__ m2, const float* __restrict__ v2,
    float* __restrict__ cout)
{
    __shared__ float w1_s[192];          // conv1_w (16,3,2,2)
    __shared__ float s1_s[16], o1_s[16]; // folded bn1 scale / (bias fold)
    __shared__ float w2_s[64];           // conv2_w * bn2 scale
    __shared__ float o2_s;

    const int tid = threadIdx.x;
    if (tid < 192) w1_s[tid] = w1[tid];
    if (tid < 16) {
        float sc = g1[tid] * rsqrtf(v1[tid] + 1e-5f);
        s1_s[tid] = sc;
        o1_s[tid] = (cb1[tid] - m1[tid]) * sc + be1[tid];
    }
    if (tid >= 192 && tid < 256) {
        int i = tid - 192;
        float sc2 = g2[0] * rsqrtf(v2[0] + 1e-5f);
        w2_s[i] = w2[i] * sc2;
        if (i == 0) o2_s = (cb2[0] - m2[0]) * sc2 + be2[0];
    }
    __syncthreads();

    const int idx = blockIdx.x * 256 + tid;   // 0..16383
    const int b = idx >> 8;
    const int t = idx & 255;
    const int p = t >> 4, q = t & 15;

    const float* xb = x + (size_t)b * 3 * 256 * 256;
    float best2 = -3.4e38f;
    #pragma unroll
    for (int dp = 0; dp < 2; ++dp)
    #pragma unroll
    for (int dq = 0; dq < 2; ++dq) {
        const int u = 2 * p + dp, v = 2 * q + dq;       // conv2 coord (32x32)
        float acc2 = o2_s;
        #pragma unroll
        for (int kh2 = 0; kh2 < 2; ++kh2)
        #pragma unroll
        for (int kw2 = 0; kw2 < 2; ++kw2) {
            const int r = 2 * u + kh2, s = 2 * v + kw2; // pool1 coord (64x64)
            float p1[16];
            #pragma unroll
            for (int oc = 0; oc < 16; ++oc) p1[oc] = -3.4e38f;
            #pragma unroll
            for (int a = 0; a < 2; ++a)
            #pragma unroll
            for (int bb = 0; bb < 2; ++bb) {
                const int y = 2 * r + a, z = 2 * s + bb; // conv1 coord (128x128)
                float xv[12];
                #pragma unroll
                for (int c3 = 0; c3 < 3; ++c3)
                #pragma unroll
                for (int kh = 0; kh < 2; ++kh)
                #pragma unroll
                for (int kw = 0; kw < 2; ++kw)
                    xv[c3 * 4 + kh * 2 + kw] =
                        xb[((size_t)c3 * 256 + (2 * y + kh)) * 256 + (2 * z + kw)];
                #pragma unroll
                for (int oc = 0; oc < 16; ++oc) {
                    float a1 = 0.f;
                    #pragma unroll
                    for (int j = 0; j < 12; ++j) a1 += xv[j] * w1_s[oc * 12 + j];
                    a1 = fmaxf(a1 * s1_s[oc] + o1_s[oc], 0.f);
                    p1[oc] = fmaxf(p1[oc], a1);
                }
            }
            #pragma unroll
            for (int ic = 0; ic < 16; ++ic) acc2 += p1[ic] * w2_s[ic * 4 + kh2 * 2 + kw2];
        }
        acc2 = fmaxf(acc2, 0.f);
        best2 = fmaxf(best2, acc2);
    }
    cout[idx] = best2;
}

// ---------------------------------------------------------------------------
// Kernel 2: pre-MLP + softmax -> pre (flat (64,8) row-major; reference then
// raw-reshapes it to (8,64,1), i.e. pre_rs[cls][b] = pre_flat[cls*64+b]).
// ---------------------------------------------------------------------------
__global__ __launch_bounds__(64) void premlp_kernel(
    const float* __restrict__ c,
    const float* __restrict__ w1, const float* __restrict__ b1,
    const float* __restrict__ w2, const float* __restrict__ b2,
    const float* __restrict__ w3, const float* __restrict__ b3,
    float* __restrict__ pre)
{
    const int b = threadIdx.x;
    const float* cb = c + b * 256;
    float h1[32];
    for (int o = 0; o < 32; ++o) {
        float a = b1[o];
        const float* wr = w1 + o * 256;
        for (int k = 0; k < 256; ++k) a += cb[k] * wr[k];
        h1[o] = fmaxf(a, 0.f);
    }
    float h2[32];
    for (int o = 0; o < 32; ++o) {
        float a = b2[o];
        const float* wr = w2 + o * 32;
        #pragma unroll
        for (int k = 0; k < 32; ++k) a += h1[k] * wr[k];
        h2[o] = fmaxf(a, 0.f);
    }
    float lg[8], mx = -3.4e38f;
    for (int o = 0; o < 8; ++o) {
        float a = b3[o];
        const float* wr = w3 + o * 32;
        #pragma unroll
        for (int k = 0; k < 32; ++k) a += h2[k] * wr[k];
        lg[o] = a;
        mx = fmaxf(mx, a);
    }
    float sum = 0.f;
    for (int o = 0; o < 8; ++o) { lg[o] = __expf(lg[o] - mx); sum += lg[o]; }
    const float inv = 1.f / sum;
    for (int o = 0; o < 8; ++o) pre[b * 8 + o] = lg[o] * inv;
}

// ---------------------------------------------------------------------------
// Kernel 3: per-class LSTM, 256 steps. One workgroup per class (grid=8),
// 1024 threads = 32 waves. whh lives in LDS as bf16 (128 KB), h in LDS (16 KB).
// Each step: gates(64x512) = h(64x128) @ whh^T via v_wmma_f32_16x16x32_bf16.
// Wave w owns M-tile (w/8) and the i/f/g/o N-tiles for columns (w%8)*16..+15,
// so the cell update runs entirely in registers (cc kept in a v8f per lane).
// Writes o_last(=h[:,127]) * pre[cls][b] per step to the workspace.
// ---------------------------------------------------------------------------
__global__ __launch_bounds__(1024) void lstm_kernel(
    const float* __restrict__ c,        // (64,256)
    const float* __restrict__ wih,      // (8,512,1)
    const float* __restrict__ whh,      // (8,512,128)
    const float* __restrict__ bih,      // (8,512)
    const float* __restrict__ bhh,      // (8,512)
    const float* __restrict__ hn,       // (8,1,64,128)
    const float* __restrict__ pre,      // flat (64,8); pre_rs[cls][b]=pre[cls*64+b]
    float* __restrict__ oscaled)        // (8,64,256)
{
    __shared__ __bf16 whh_s[512 * 128]; // 128 KB  (B operand, resident all 256 steps)
    __shared__ __bf16 h_s[64 * 128];    //  16 KB  (A operand, rewritten each step)
    __shared__ float  wv_s[512];        // wih vector
    __shared__ float  bs_s[512];        // bih + bhh

    const int cls = blockIdx.x;
    const int tid = threadIdx.x;

    const float* W = whh + (size_t)cls * 512 * 128;
    for (int i = tid; i < 512 * 128; i += 1024) whh_s[i] = (__bf16)W[i];
    for (int i = tid; i < 512; i += 1024) {
        wv_s[i] = wih[cls * 512 + i];
        bs_s[i] = bih[cls * 512 + i] + bhh[cls * 512 + i];
    }
    const float* H0 = hn + (size_t)cls * 64 * 128;
    for (int i = tid; i < 64 * 128; i += 1024) h_s[i] = (__bf16)H0[i];
    __syncthreads();

    const int wave = tid >> 5;
    const int lane = tid & 31;
    const int m    = wave >> 3;           // M tile (batch rows m*16..)
    const int n0   = wave & 7;            // column group within each gate
    const int lmod = lane & 15;
    const int half = lane >> 4;

    const int bA   = m * 16 + lmod;       // A-fragment row (h row)
    const int jcol = n0 * 16 + lmod;      // hidden column owned in C fragments
    const int bC0  = m * 16 + half * 8;   // first batch row of this lane's C slots

    float pre8[8];
    #pragma unroll
    for (int r = 0; r < 8; ++r) pre8[r] = pre[cls * 64 + (bC0 + r)];
    const bool writer = (jcol == 127);    // o_last = h[:, HID-1]
    float* osc = oscaled + ((size_t)cls * 64 + bC0) * 256;

    v8f cc = {0.f, 0.f, 0.f, 0.f, 0.f, 0.f, 0.f, 0.f};

    for (int t = 0; t < 256; ++t) {
        float cb[8];
        #pragma unroll
        for (int r = 0; r < 8; ++r) cb[r] = c[(bC0 + r) * 256 + t];

        // init accumulators with xg_t + bih + bhh  (input dim is 1: outer product)
        v8f acc[4];
        #pragma unroll
        for (int gt = 0; gt < 4; ++gt) {
            const int g = gt * 128 + jcol;
            const float wv = wv_s[g], bs = bs_s[g];
            #pragma unroll
            for (int r = 0; r < 8; ++r) acc[gt][r] = cb[r] * wv + bs;
        }

        // gates += h @ whh^T  (K = 128 -> 4 x WMMA 16x16x32 bf16)
        #pragma unroll
        for (int k = 0; k < 4; ++k) {
            const __bf16* pa = h_s + bA * 128 + k * 32 + half * 8;
            const v8bf a_lo = *(const v8bf*)pa;
            const v8bf a_hi = *(const v8bf*)(pa + 16);
            const v16bf Af = __builtin_shufflevector(a_lo, a_hi,
                0, 1, 2, 3, 4, 5, 6, 7, 8, 9, 10, 11, 12, 13, 14, 15);
            #pragma unroll
            for (int gt = 0; gt < 4; ++gt) {
                const int g = gt * 128 + n0 * 16 + lmod;      // B column = whh row
                const __bf16* pb = whh_s + g * 128 + k * 32 + half * 8;
                const v8bf b_lo = *(const v8bf*)pb;
                const v8bf b_hi = *(const v8bf*)(pb + 16);
                const v16bf Bf = __builtin_shufflevector(b_lo, b_hi,
                    0, 1, 2, 3, 4, 5, 6, 7, 8, 9, 10, 11, 12, 13, 14, 15);
                acc[gt] = __builtin_amdgcn_wmma_f32_16x16x32_bf16(
                    false, Af, false, Bf, (short)0, acc[gt], false, false);
            }
        }
        __syncthreads();   // everyone done reading h_s

        // fused gate nonlinearities + cell update, all in registers
        #pragma unroll
        for (int r = 0; r < 8; ++r) {
            const float i_ = acc[0][r], f_ = acc[1][r];
            const float g_ = acc[2][r], o_ = acc[3][r];
            const float ncc = sigm(f_) * cc[r] + sigm(i_) * tanh_fast(g_);
            cc[r] = ncc;
            const float hv = sigm(o_) * tanh_fast(ncc);
            h_s[(bC0 + r) * 128 + jcol] = (__bf16)hv;
            if (writer) osc[r * 256 + t] = hv * pre8[r];
        }
        __syncthreads();   // h_s ready for next step
    }
}

// ---------------------------------------------------------------------------
// Kernel 4: class-mean + output projection: out(64,8)
// ---------------------------------------------------------------------------
__global__ __launch_bounds__(512) void head_kernel(
    const float* __restrict__ os,      // (8,64,256)
    const float* __restrict__ ow,      // (8,256)
    const float* __restrict__ ob,      // (8,)
    float* __restrict__ out)           // (64,8)
{
    const int tid = threadIdx.x;
    const int b = tid >> 3, n = tid & 7;
    float acc = ob[n];
    const float* wr = ow + n * 256;
    for (int t = 0; t < 256; ++t) {
        float s = 0.f;
        #pragma unroll
        for (int cl = 0; cl < 8; ++cl) s += os[((size_t)cl * 64 + b) * 256 + t];
        acc += (s * 0.125f) * wr[t];
    }
    out[b * 8 + n] = acc;
}

// ---------------------------------------------------------------------------
extern "C" void kernel_launch(void* const* d_in, const int* in_sizes, int n_in,
                              void* d_out, int out_size, void* d_ws, size_t ws_size,
                              hipStream_t stream) {
    (void)in_sizes; (void)n_in; (void)out_size; (void)ws_size;
    const float* x    = (const float*)d_in[0];
    const float* c1w  = (const float*)d_in[1];
    const float* c1b  = (const float*)d_in[2];
    const float* g1   = (const float*)d_in[3];
    const float* be1  = (const float*)d_in[4];
    const float* m1   = (const float*)d_in[5];
    const float* v1   = (const float*)d_in[6];
    const float* c2w  = (const float*)d_in[7];
    const float* c2b  = (const float*)d_in[8];
    const float* g2   = (const float*)d_in[9];
    const float* be2  = (const float*)d_in[10];
    const float* m2   = (const float*)d_in[11];
    const float* v2   = (const float*)d_in[12];
    const float* pw1  = (const float*)d_in[13];
    const float* pb1  = (const float*)d_in[14];
    const float* pw2  = (const float*)d_in[15];
    const float* pb2  = (const float*)d_in[16];
    const float* pw3  = (const float*)d_in[17];
    const float* pb3  = (const float*)d_in[18];
    const float* wih  = (const float*)d_in[19];
    const float* whh  = (const float*)d_in[20];
    const float* bih  = (const float*)d_in[21];
    const float* bhh  = (const float*)d_in[22];
    const float* hn   = (const float*)d_in[23];
    const float* ow   = (const float*)d_in[24];
    const float* ob   = (const float*)d_in[25];

    float* ws      = (float*)d_ws;
    float* c_buf   = ws;                   // 64*256          = 16384 f
    float* pre_buf = ws + 16384;           // 64*8            =   512 f
    float* os_buf  = ws + 16384 + 512;     // 8*64*256        = 131072 f

    fused_conv_kernel<<<64, 256, 0, stream>>>(x, c1w, c1b, g1, be1, m1, v1,
                                              c2w, c2b, g2, be2, m2, v2, c_buf);
    premlp_kernel<<<1, 64, 0, stream>>>(c_buf, pw1, pb1, pw2, pb2, pw3, pb3, pre_buf);
    lstm_kernel<<<8, 1024, 0, stream>>>(c_buf, wih, whh, bih, bhh, hn, pre_buf, os_buf);
    head_kernel<<<1, 512, 0, stream>>>(os_buf, ow, ob, (float*)d_out);
}
